// GPT_77756087927496
// MI455X (gfx1250) — matrix operand order repaired and checked by reading, not compiled, not run on hardware
//
#include <hip/hip_runtime.h>
#include <math.h>

// ---------------- CDNA5 types ----------------
typedef __bf16 v16bf __attribute__((ext_vector_type(16)));
typedef __bf16 v8bf  __attribute__((ext_vector_type(8)));
typedef float  v8f   __attribute__((ext_vector_type(8)));
typedef float  v8fv  __attribute__((ext_vector_type(8)));

// ---------------- problem constants ----------------
#define BB   8
#define TTQ  256
#define TSF  512
#define CCH  512
#define NH   8
#define HDD  64
#define NLAY 10

// ---------------- helpers ----------------
__device__ __forceinline__ v8bf cvt8(const float* __restrict__ p) {
  // hardware packed f32->bf16 conversion via vector fptrunc
  return __builtin_convertvector(*(const v8fv*)p, v8bf);
}
__device__ __forceinline__ float wred_sum(float v) {
#pragma unroll
  for (int o = 16; o > 0; o >>= 1) v += __shfl_xor(v, o, 32);
  return v;
}
__device__ __forceinline__ int wred_sum_i(int v) {
#pragma unroll
  for (int o = 16; o > 0; o >>= 1) v += __shfl_xor(v, o, 32);
  return v;
}
__device__ __forceinline__ float wred_max(float v) {
#pragma unroll
  for (int o = 16; o > 0; o >>= 1) v = fmaxf(v, __shfl_xor(v, o, 32));
  return v;
}

// ---------------- LayerNorm: one block (128 thr) per row of C=512 ----------------
__global__ __launch_bounds__(128) void ln_kernel(const float* __restrict__ x,
                                                 const float* __restrict__ g,
                                                 const float* __restrict__ bt,
                                                 float* __restrict__ out) {
  const int row = blockIdx.x, tid = threadIdx.x;
  const float* xr = x + (size_t)row * CCH;
  float loc[4];
  float s = 0.f;
#pragma unroll
  for (int i = 0; i < 4; ++i) { loc[i] = xr[tid + i * 128]; s += loc[i]; }
  __shared__ float red[128];
  red[tid] = s; __syncthreads();
  for (int o = 64; o > 0; o >>= 1) { if (tid < o) red[tid] += red[tid + o]; __syncthreads(); }
  const float mu = red[0] * (1.0f / CCH);
  __syncthreads();
  float vs = 0.f;
#pragma unroll
  for (int i = 0; i < 4; ++i) { float d = loc[i] - mu; vs += d * d; }
  red[tid] = vs; __syncthreads();
  for (int o = 64; o > 0; o >>= 1) { if (tid < o) red[tid] += red[tid + o]; __syncthreads(); }
  const float rstd = rsqrtf(red[0] * (1.0f / CCH) + 1e-5f);
  float* orow = out + (size_t)row * CCH;
#pragma unroll
  for (int i = 0; i < 4; ++i) {
    int c = tid + i * 128;
    orow[c] = (loc[i] - mu) * rstd * g[c] + bt[c];
  }
}

// ---------------- staged tile loader (global f32 -> LDS bf16) ----------------
// As: [128 rows][40] (K contiguous, 80B row stride, 16B aligned chunks)
// Bs: [64 cols][40]  (transposed: K contiguous per output column)
__device__ __forceinline__ void stage_tile(const float* __restrict__ A,
                                           const float* __restrict__ W,
                                           int K, int N, int tileM, int tileN, int kt,
                                           __bf16 (*__restrict__ as)[40],
                                           __bf16 (*__restrict__ bs)[40],
                                           int arow, int acol0, int brow, int bcol0) {
  const float* ap = A + (size_t)(tileM + arow) * K + (kt * 32 + acol0);
  const float* bp = W + (size_t)(kt * 32 + brow) * N + (tileN + bcol0);
  v8bf alo = cvt8(ap);          // global_load_b128 x2 -> v_cvt pk bf16
  v8bf ahi = cvt8(ap + 8);
  v8bf bb  = cvt8(bp);
  *(v8bf*)&as[arow][acol0]     = alo;   // ds_store_b128
  *(v8bf*)&as[arow][acol0 + 8] = ahi;   // ds_store_b128
#pragma unroll
  for (int i = 0; i < 8; ++i) bs[bcol0 + i][brow] = bb[i];   // transposed scatter (b16)
}

// ---------------- bf16 WMMA GEMM: out[M,N] = act(A[M,K] @ W[K,N] + bias) (+residual) ----
// Block 256 thr = 8 waves; tile 128(M) x 64(N); wave = 32x32 via 2x2 wmma accumulators.
// Double-buffered LDS pipeline: global loads of tile k+1 overlap WMMA of tile k.
template <int ACT, int RES>
__global__ __launch_bounds__(256) void gemm_bf16_wmma(const float* __restrict__ A,
                                                      const float* __restrict__ W,
                                                      const float* __restrict__ bias,
                                                      float* __restrict__ out,
                                                      const float* __restrict__ residual,
                                                      int M, int N, int K) {
  __shared__ __bf16 As[2][128][40];   // 20 KB
  __shared__ __bf16 Bs[2][64][40];    // 10 KB
  const int tid = threadIdx.x;
  const int lane = tid & 31, wid = tid >> 5;
  const int waveM = wid & 3, waveN = wid >> 2;
  const int l15 = lane & 15, lhi = lane >> 4;
  const int tileM = blockIdx.y * 128, tileN = blockIdx.x * 64;

  const int arow = tid >> 1, acol0 = (tid & 1) << 4;   // 16 A elems / thread
  const int brow = tid >> 3, bcol0 = (tid & 7) << 3;   // 8  B elems / thread

  v8f z = {};
  v8f acc[2][2];
  acc[0][0] = z; acc[0][1] = z; acc[1][0] = z; acc[1][1] = z;

  const int rA0 = waveM * 32 + l15;
  const int cB0 = waveN * 32 + l15;
  const int nkt = K >> 5;

  stage_tile(A, W, K, N, tileM, tileN, 0, As[0], Bs[0], arow, acol0, brow, bcol0);

  for (int kt = 0; kt < nkt; ++kt) {
    __syncthreads();
    if (kt + 1 < nkt)
      stage_tile(A, W, K, N, tileM, tileN, kt + 1, As[(kt + 1) & 1], Bs[(kt + 1) & 1],
                 arow, acol0, brow, bcol0);
    if (kt + 2 < nkt) {   // global_prefetch_b8, near-temporal
      __builtin_prefetch(A + (size_t)(tileM + arow) * K + ((kt + 2) * 32 + acol0), 0, 3);
      __builtin_prefetch(W + (size_t)((kt + 2) * 32 + brow) * N + (tileN + bcol0), 0, 3);
    }
    const int cb = kt & 1;

    // A fragment (ISA 16-bit A 16x32 layout): lane half selects K base (+8);
    // per lane: chunk [lhi*8 .. +7] and [16+lhi*8 .. +7]  -> two ds_load_b128
    const __bf16* pa0 = &As[cb][rA0][lhi * 8];
    const __bf16* pa1 = &As[cb][rA0 + 16][lhi * 8];
    v8bf a0lo = *(const v8bf*)pa0,        a0hi = *(const v8bf*)(pa0 + 16);
    v8bf a1lo = *(const v8bf*)pa1,        a1hi = *(const v8bf*)(pa1 + 16);
    // B fragment (32x16): same K order as A: chunk [lhi*8 .. +7] and [16+lhi*8 .. +7] in Bs[col][]
    const __bf16* pb0 = &Bs[cb][cB0][lhi * 8];
    const __bf16* pb1 = &Bs[cb][cB0 + 16][lhi * 8];
    v8bf b0lo = *(const v8bf*)pb0,        b0hi = *(const v8bf*)(pb0 + 16);
    v8bf b1lo = *(const v8bf*)pb1,        b1hi = *(const v8bf*)(pb1 + 16);

    v16bf a0 = __builtin_shufflevector(a0lo, a0hi, 0,1,2,3,4,5,6,7,8,9,10,11,12,13,14,15);
    v16bf a1 = __builtin_shufflevector(a1lo, a1hi, 0,1,2,3,4,5,6,7,8,9,10,11,12,13,14,15);
    v16bf b0 = __builtin_shufflevector(b0lo, b0hi, 0,1,2,3,4,5,6,7,8,9,10,11,12,13,14,15);
    v16bf b1 = __builtin_shufflevector(b1lo, b1hi, 0,1,2,3,4,5,6,7,8,9,10,11,12,13,14,15);

    acc[0][0] = __builtin_amdgcn_wmma_f32_16x16x32_bf16(false, a0, false, b0, (short)0, acc[0][0], false, false);
    acc[0][1] = __builtin_amdgcn_wmma_f32_16x16x32_bf16(false, a0, false, b1, (short)0, acc[0][1], false, false);
    acc[1][0] = __builtin_amdgcn_wmma_f32_16x16x32_bf16(false, a1, false, b0, (short)0, acc[1][0], false, false);
    acc[1][1] = __builtin_amdgcn_wmma_f32_16x16x32_bf16(false, a1, false, b1, (short)0, acc[1][1], false, false);
  }

  // Epilogue: C/D layout -> VGPR r: lane<16 M=r, lane>=16 M=8+r; N = lane&15
#pragma unroll
  for (int mt = 0; mt < 2; ++mt) {
#pragma unroll
    for (int nt = 0; nt < 2; ++nt) {
      const int gn = tileN + waveN * 32 + nt * 16 + l15;
      const float bv = bias[gn];
#pragma unroll
      for (int r = 0; r < 8; ++r) {
        const int gm = tileM + waveM * 32 + mt * 16 + (lhi ? 8 + r : r);
        float v = acc[mt][nt][r] + bv;
        if (ACT) v = 0.5f * v * (1.0f + erff(v * 0.70710678118654752f));  // exact GELU
        if (RES) v += residual[(size_t)gm * N + gn];
        out[(size_t)gm * N + gn] = v;
      }
    }
  }
}

// ---------------- Attention: one wave per query row ----------------
// Q,K,V,Y layouts: [B, T, C] with head-interleaved features (feature = h*64 + d).
__global__ __launch_bounds__(256) void attn_kernel(const float* __restrict__ Q,
                                                   const float* __restrict__ K,
                                                   const float* __restrict__ V,
                                                   float* __restrict__ Y,
                                                   int Ts, int causal, int topk) {
  const int tid = threadIdx.x, w = tid >> 5, lane = tid & 31;
  const int tg = blockIdx.x % (TTQ / 8);
  const int h  = (blockIdx.x / (TTQ / 8)) % NH;
  const int b  = blockIdx.x / ((TTQ / 8) * NH);
  const int t  = tg * 8 + w;

  __shared__ float sq[8][64];
  __shared__ float sw[8][512];

  const float* qr = Q + ((size_t)(b * TTQ + t)) * CCH + h * HDD;
  sq[w][lane] = qr[lane];
  sq[w][lane + 32] = qr[lane + 32];
  __syncthreads();

  const int nk = Ts >> 5;            // keys per lane (8 or 16)
  float sc[16];
  const float scale = 0.125f;        // 1/sqrt(64)
  for (int i = 0; i < nk; ++i) {
    const int j = lane + (i << 5);
    const float* kr = K + ((size_t)(b * Ts + j)) * CCH + h * HDD;
    float d = 0.f;
#pragma unroll
    for (int c = 0; c < HDD; ++c) d += sq[w][c] * kr[c];
    d *= scale;
    if (causal && j > t) d = -__builtin_inff();
    sc[i] = d;
  }

  // top-k threshold: radix select of k-th largest over order-preserving uint map
  if (topk > 0 && topk < Ts) {
    unsigned prefix = 0u;
    for (int bit = 31; bit >= 0; --bit) {
      const unsigned cand = prefix | (1u << bit);
      int cnt = 0;
      for (int i = 0; i < nk; ++i) {
        unsigned u = __float_as_uint(sc[i]);
        u = (u & 0x80000000u) ? ~u : (u | 0x80000000u);
        cnt += (u >= cand) ? 1 : 0;
      }
      cnt = wred_sum_i(cnt);
      if (cnt >= topk) prefix = cand;
    }
    for (int i = 0; i < nk; ++i) {
      unsigned u = __float_as_uint(sc[i]);
      u = (u & 0x80000000u) ? ~u : (u | 0x80000000u);
      if (u < prefix) sc[i] = -__builtin_inff();   // att < kth-largest -> -inf (ties kept)
    }
  }

  // softmax across the row (wave reduction)
  float m = -__builtin_inff();
  for (int i = 0; i < nk; ++i) m = fmaxf(m, sc[i]);
  m = wred_max(m);
  float s = 0.f;
  for (int i = 0; i < nk; ++i) { float e = __expf(sc[i] - m); s += e; sc[i] = e; }
  s = wred_sum(s);
  const float inv = 1.0f / s;
  for (int i = 0; i < nk; ++i) sw[w][lane + (i << 5)] = sc[i] * inv;
  __syncthreads();

  // y[d] = sum_j w[j] * V[b][j][h*64+d]; lanes cover d = lane, lane+32 (coalesced V reads)
  for (int d0 = lane; d0 < HDD; d0 += 32) {
    float acc = 0.f;
    const float* vp = V + ((size_t)b * Ts) * CCH + h * HDD + d0;
#pragma unroll 4
    for (int j = 0; j < Ts; ++j) acc += sw[w][j] * vp[(size_t)j * CCH];
    Y[((size_t)(b * TTQ + t)) * CCH + h * HDD + d0] = acc;
  }
}

// ---------------- orchestration ----------------
extern "C" void kernel_launch(void* const* d_in, const int* in_sizes, int n_in,
                              void* d_out, int out_size, void* d_ws, size_t ws_size,
                              hipStream_t stream) {
  const float* id_x    = (const float*)d_in[0];
  const float* id_prev = (const float*)d_in[1];
  const float* frames  = (const float*)d_in[2];
  const float* Wq = (const float*)d_in[3];  const float* bq = (const float*)d_in[4];
  const float* Wk = (const float*)d_in[5];  const float* bk = (const float*)d_in[6];
  const float* Wv = (const float*)d_in[7];  const float* bv = (const float*)d_in[8];
  const float* Wo = (const float*)d_in[9];  const float* bo = (const float*)d_in[10];
  const float* W1 = (const float*)d_in[11]; const float* b1 = (const float*)d_in[12];
  const float* W2 = (const float*)d_in[13]; const float* b2 = (const float*)d_in[14];
  const float* l1g = (const float*)d_in[15]; const float* l1b = (const float*)d_in[16];
  const float* l2g = (const float*)d_in[17]; const float* l2b = (const float*)d_in[18];
  const float* lfg = (const float*)d_in[19]; const float* lfb = (const float*)d_in[20];

  const int Mq  = BB * TTQ;                 // 2048
  const size_t nX = (size_t)Mq * CCH;       // 1M floats

  float* ws = (float*)d_ws;
  size_t o = 0;
  float* X  = ws + o; o += nX;                       // running activations
  float* Hb = ws + o; o += nX;                       // ln output / scratch
  float* Qb = ws + o; o += nX;
  float* Kb = ws + o; o += (size_t)BB * TSF * CCH;   // max kv rows
  float* Vb = ws + o; o += (size_t)BB * TSF * CCH;
  float* Yb = ws + o; o += nX;
  float* T1 = ws + o; o += (size_t)Mq * 4 * CCH;     // MLP hidden

  hipMemcpyAsync(X, id_x, nX * sizeof(float), hipMemcpyDeviceToDevice, stream);

  // layer schedule: 4x stimulus cross-attn (frames, topk=64), 2x causal self,
  // 2x history cross-attn (id_prev), 2x causal self
  const int src[NLAY]    = {0, 0, 0, 0, 1, 1, 2, 2, 1, 1};
  const int causal[NLAY] = {0, 0, 0, 0, 1, 1, 0, 0, 1, 1};
  const int topk[NLAY]   = {64, 64, 64, 64, 0, 0, 0, 0, 0, 0};

  const size_t CxC = (size_t)CCH * CCH;
  const size_t Cx4C = (size_t)CCH * 4 * CCH;

  for (int i = 0; i < NLAY; ++i) {
    const float* kv = (src[i] == 0) ? frames : (src[i] == 1 ? X : id_prev);
    const int Ts  = (src[i] == 0) ? TSF : TTQ;
    const int Mkv = BB * Ts;

    // h = ln1(x)
    ln_kernel<<<Mq, 128, 0, stream>>>(X, l1g + i * CCH, l1b + i * CCH, Hb);
    // Q = h@Wq+bq ; K = kv@Wk+bk ; V = kv@Wv+bv
    gemm_bf16_wmma<0, 0><<<dim3(CCH / 64, Mq / 128), 256, 0, stream>>>(
        Hb, Wq + i * CxC, bq + i * CCH, Qb, nullptr, Mq, CCH, CCH);
    gemm_bf16_wmma<0, 0><<<dim3(CCH / 64, Mkv / 128), 256, 0, stream>>>(
        kv, Wk + i * CxC, bk + i * CCH, Kb, nullptr, Mkv, CCH, CCH);
    gemm_bf16_wmma<0, 0><<<dim3(CCH / 64, Mkv / 128), 256, 0, stream>>>(
        kv, Wv + i * CxC, bv + i * CCH, Vb, nullptr, Mkv, CCH, CCH);
    // attention (causal mask / top-k / softmax / @V), heads merged in Yb
    attn_kernel<<<BB * NH * (TTQ / 8), 256, 0, stream>>>(Qb, Kb, Vb, Yb, Ts, causal[i], topk[i]);
    // x = x + y@Wo + bo   (residual fused, in place on X)
    gemm_bf16_wmma<0, 1><<<dim3(CCH / 64, Mq / 128), 256, 0, stream>>>(
        Yb, Wo + i * CxC, bo + i * CCH, X, X, Mq, CCH, CCH);
    // h2 = ln2(x)
    ln_kernel<<<Mq, 128, 0, stream>>>(X, l2g + i * CCH, l2b + i * CCH, Hb);
    // t1 = gelu(h2@W1 + b1)
    gemm_bf16_wmma<1, 0><<<dim3(4 * CCH / 64, Mq / 128), 256, 0, stream>>>(
        Hb, W1 + i * Cx4C, b1 + i * 4 * CCH, T1, nullptr, Mq, 4 * CCH, CCH);
    // x = x + t1@W2 + b2
    gemm_bf16_wmma<0, 1><<<dim3(CCH / 64, Mq / 128), 256, 0, stream>>>(
        T1, W2 + i * Cx4C, b2 + i * CCH, X, X, Mq, CCH, 4 * CCH);
    // x = ln_f(x), in place
    ln_kernel<<<Mq, 128, 0, stream>>>(X, lfg + i * CCH, lfb + i * CCH, X);
  }

  hipMemcpyAsync(d_out, X, nX * sizeof(float), hipMemcpyDeviceToDevice, stream);
}
